// GraphSAGE_4Layer_33328946217667
// MI455X (gfx1250) — matrix operand order repaired and checked by reading, not compiled, not run on hardware
//
#include <hip/hip_runtime.h>
#include <hip/hip_bf16.h>

#define N_NODES  50000
#define N_EDGES  600000
#define N_GRAPHS 256
#define C        128    // IN_C == HID
#define OUT_C    2
#define WMAT_ELEMS (C * C)   // 16384

typedef __attribute__((ext_vector_type(16))) _Float16 v16h;
typedef __attribute__((ext_vector_type(8)))  _Float16 v8h;
typedef __attribute__((ext_vector_type(4)))  _Float16 v4h;
typedef __attribute__((ext_vector_type(2)))  _Float16 v2h;
typedef __attribute__((ext_vector_type(8)))  float    v8f;

// hardware packed-f16 atomic add (GLOBAL_ATOMIC_PK_ADD_F16, no return -> STOREcnt;
// drained by s_endpgm implicit wait-idle, cross-kernel ordering via kernel boundary)
__device__ __forceinline__ void atomic_pk_add_f16(_Float16* addr, unsigned int packed) {
    asm volatile("global_atomic_pk_add_f16 %0, %1, off"
                 :: "v"(addr), "v"(packed) : "memory");
}

// ---------------- utility: zero a buffer of 32-bit words (grid-stride) ----------------
__global__ void zero_kernel(float* __restrict__ p, int n) {
    int i = blockIdx.x * blockDim.x + threadIdx.x;
    int stride = gridDim.x * blockDim.x;
    for (; i < n; i += stride) p[i] = 0.0f;
}

// ---------------- f32 -> f16 conversion, 2 elements per thread ----------------
__global__ void cvt_f16_kernel(const float* __restrict__ in, _Float16* __restrict__ out, int n2) {
    int i = blockIdx.x * blockDim.x + threadIdx.x;
    if (i >= n2) return;
    const float2 v = ((const float2*)in)[i];
    v2h o;
    o[0] = (_Float16)v.x;
    o[1] = (_Float16)v.y;
    ((v2h*)out)[i] = o;
}

// ---------------- degree count + inversion ----------------
__global__ void deg_kernel(const int* __restrict__ tgt, float* __restrict__ deg) {
    int e = blockIdx.x * blockDim.x + threadIdx.x;
    if (e < N_EDGES) unsafeAtomicAdd(&deg[tgt[e]], 1.0f);
}

__global__ void deg_inv_kernel(float* __restrict__ deg) {
    int i = blockIdx.x * blockDim.x + threadIdx.x;
    if (i < N_NODES) deg[i] = 1.0f / fmaxf(deg[i], 1.0f);
}

// ---------------- edge scatter: wave32 per edge, 4 f16 channels per lane,
// ---------------- two packed-f16 hardware atomics per lane ----------------
__global__ void scatter_kernel(const _Float16* __restrict__ x,
                               const int*      __restrict__ src,
                               const int*      __restrict__ tgt,
                               _Float16*       __restrict__ agg) {
    int e = (blockIdx.x * blockDim.x + threadIdx.x) >> 5;
    if (e >= N_EDGES) return;
    int lane = threadIdx.x & 31;
    int s = src[e];
    int t = tgt[e];
    const uint2 v = *(const uint2*)(x + (size_t)s * C + lane * 4);
    _Float16* dst = agg + (size_t)t * C + lane * 4;
    atomic_pk_add_f16(dst,     v.x);
    atomic_pk_add_f16(dst + 2, v.y);
}

// ---------------- fused SAGEConv GEMM (all-f16 operands, f32 accumulate):
// out = relu(agg*deg_inv @ w_l^T + b_l + x @ w_r^T), one wave per 16x16 tile.
__global__ void __launch_bounds__(256)
sage_gemm_kernel(const _Float16* __restrict__ xin,
                 const _Float16* __restrict__ agg,
                 const float*    __restrict__ deg_inv,
                 const _Float16* __restrict__ w_l,
                 const float*    __restrict__ b_l,
                 const _Float16* __restrict__ w_r,
                 _Float16*       __restrict__ out) {
    const int wave   = threadIdx.x >> 5;
    const int lane   = threadIdx.x & 31;
    const int gid    = blockIdx.x * 8 + wave;
    const int tile_m = gid >> 3;      // 0..3124
    const int tile_n = gid & 7;       // 0..7
    const int half   = lane >> 4;     // 0 or 1
    const int l16    = lane & 15;

    const int node = tile_m * 16 + l16;   // A-row owned by this lane
    const int col  = tile_n * 16 + l16;   // B-column / C-column owned by this lane
    const _Float16 sh = (_Float16)deg_inv[node];
    const _Float16* aggrow = agg + (size_t)node * C;
    const _Float16* xrow   = xin + (size_t)node * C;

    v8f acc = {};
    #pragma unroll
    for (int kb = 0; kb < C; kb += 32) {
        const int k0 = kb + half * 16;
        // A operands: lane holds row l16; elems 0-7 <- k0.., elems 8-15 <- k0+8..
        v8h alo = *(const v8h*)(aggrow + k0);
        v8h ahi = *(const v8h*)(aggrow + k0 + 8);
        alo *= sh;                       // fold deg_inv (v_pk_mul_f16)
        ahi *= sh;
        v16h aA = __builtin_shufflevector(alo, ahi,
                    0,1,2,3,4,5,6,7,8,9,10,11,12,13,14,15);
        v8h xlo = *(const v8h*)(xrow + k0);
        v8h xhi = *(const v8h*)(xrow + k0 + 8);
        v16h aX = __builtin_shufflevector(xlo, xhi,
                    0,1,2,3,4,5,6,7,8,9,10,11,12,13,14,15);
        // B operands: lane holds column `col`, 16 consecutive k values (32B load)
        const int kw = kb + half * 16;
        const v16h bL = *(const v16h*)(w_l + (size_t)col * C + kw);
        const v16h bR = *(const v16h*)(w_r + (size_t)col * C + kw);

        acc = __builtin_amdgcn_wmma_f32_16x16x32_f16(false, aA, false, bL,
                                                     (short)0, acc, false, false);
        acc = __builtin_amdgcn_wmma_f32_16x16x32_f16(false, aX, false, bR,
                                                     (short)0, acc, false, false);
    }

    // epilogue: VGPR i -> row (half*8 + i), col = l16; bias + relu, store f16
    const float bias = b_l[col];
    #pragma unroll
    for (int i = 0; i < 8; ++i) {
        const int row = tile_m * 16 + half * 8 + i;
        float v = acc[i] + bias;
        v = v > 0.0f ? v : 0.0f;
        out[(size_t)row * C + col] = (_Float16)v;
    }
}

// ---------------- graph pooling: wave32 per node, 4 f16 channels per lane ----------------
__global__ void pool_kernel(const _Float16* __restrict__ h,
                            const int*      __restrict__ batch,
                            float*          __restrict__ g_sum,
                            float*          __restrict__ g_cnt) {
    int n = (blockIdx.x * blockDim.x + threadIdx.x) >> 5;
    if (n >= N_NODES) return;
    int lane = threadIdx.x & 31;
    int b = batch[n];
    const v4h v = *(const v4h*)(h + (size_t)n * C + lane * 4);
    float* dst = g_sum + (size_t)b * C + lane * 4;
    unsafeAtomicAdd(dst + 0, (float)v[0]);
    unsafeAtomicAdd(dst + 1, (float)v[1]);
    unsafeAtomicAdd(dst + 2, (float)v[2]);
    unsafeAtomicAdd(dst + 3, (float)v[3]);
    if (lane == 0) unsafeAtomicAdd(&g_cnt[b], 1.0f);
}

// ---------------- classifier: one block (64 threads) per graph ----------------
__global__ void cls_kernel(const _Float16* __restrict__ h,
                           const int*      __restrict__ root_idx,
                           const float*    __restrict__ g_sum,
                           const float*    __restrict__ g_cnt,
                           const float*    __restrict__ w_cls,
                           const float*    __restrict__ b_cls,
                           float*          __restrict__ out) {
    __shared__ float red[OUT_C][64];
    const int g = blockIdx.x;
    const int t = threadIdx.x;  // 0..63
    const int root = root_idx[g];
    const float inv = 1.0f / fmaxf(g_cnt[g], 1.0f);
    float s0 = 0.0f, s1 = 0.0f;
    for (int k = t; k < 2 * C; k += 64) {
        const float c = (k < C) ? (float)h[(size_t)root * C + k]
                                : g_sum[(size_t)g * C + (k - C)] * inv;
        s0 += w_cls[k] * c;            // w_cls[0][k]
        s1 += w_cls[2 * C + k] * c;    // w_cls[1][k]
    }
    red[0][t] = s0;
    red[1][t] = s1;
    __syncthreads();
    if (t < OUT_C) {
        float s = 0.0f;
        for (int i = 0; i < 64; ++i) s += red[t][i];
        out[g * OUT_C + t] = s + b_cls[t];
    }
}

// ---------------- launch ----------------
extern "C" void kernel_launch(void* const* d_in, const int* in_sizes, int n_in,
                              void* d_out, int out_size, void* d_ws, size_t ws_size,
                              hipStream_t stream) {
    (void)in_sizes; (void)n_in; (void)out_size; (void)ws_size;

    const float* x     = (const float*)d_in[0];
    const int*   ei    = (const int*)d_in[1];
    const int*   src   = ei;             // edge_index[0]
    const int*   tgt   = ei + N_EDGES;   // edge_index[1]
    const int*   root  = (const int*)d_in[2];
    const int*   batch = (const int*)d_in[3];
    const float* wf[8] = {(const float*)d_in[4],  (const float*)d_in[6],   // w1_l, w1_r
                          (const float*)d_in[7],  (const float*)d_in[9],   // w2_l, w2_r
                          (const float*)d_in[10], (const float*)d_in[12],  // w3_l, w3_r
                          (const float*)d_in[13], (const float*)d_in[15]}; // w4_l, w4_r
    const float* bl[4] = {(const float*)d_in[5],  (const float*)d_in[8],
                          (const float*)d_in[11], (const float*)d_in[14]};
    const float* w_cls = (const float*)d_in[16];
    const float* b_cls = (const float*)d_in[17];

    // workspace layout
    float* wsf = (float*)d_ws;
    float* deg   = wsf;                       // N_NODES f32 (deg -> deg_inv in place)
    float* g_sum = deg + N_NODES;             // N_GRAPHS * C f32
    float* g_cnt = g_sum + N_GRAPHS * C;      // N_GRAPHS f32
    _Float16* h16 = (_Float16*)(g_cnt + N_GRAPHS);
    _Float16* xh  = h16;                                   // N_NODES * C f16
    _Float16* agg = xh  + (size_t)N_NODES * C;             // N_NODES * C f16
    _Float16* hA  = agg + (size_t)N_NODES * C;             // N_NODES * C f16
    _Float16* hB  = hA  + (size_t)N_NODES * C;             // N_NODES * C f16
    _Float16* wh  = hB  + (size_t)N_NODES * C;             // 8 * 16384 f16

    const int ZB = 256;
    // degree
    zero_kernel<<<256, ZB, 0, stream>>>(deg, N_NODES);
    deg_kernel<<<(N_EDGES + ZB - 1) / ZB, ZB, 0, stream>>>(tgt, deg);
    deg_inv_kernel<<<(N_NODES + ZB - 1) / ZB, ZB, 0, stream>>>(deg);

    // one-time f32 -> f16 conversions (input features + all 8 weight matrices)
    const int xn2 = N_NODES * C / 2;
    cvt_f16_kernel<<<(xn2 + ZB - 1) / ZB, ZB, 0, stream>>>(x, xh, xn2);
    const int wn2 = WMAT_ELEMS / 2;
    for (int m = 0; m < 8; ++m)
        cvt_f16_kernel<<<(wn2 + ZB - 1) / ZB, ZB, 0, stream>>>(wf[m], wh + (size_t)m * WMAT_ELEMS, wn2);

    const int scatter_blocks = (N_EDGES * 32 + ZB - 1) / ZB;  // 75000
    const int gemm_blocks    = N_NODES / 16;                  // 3125 blocks x 8 waves
    _Float16* bufs[2] = {hA, hB};
    const _Float16* cur = xh;
    for (int l = 0; l < 4; ++l) {
        zero_kernel<<<2048, ZB, 0, stream>>>((float*)agg, N_NODES * C / 2);  // f16 pairs as u32
        scatter_kernel<<<scatter_blocks, ZB, 0, stream>>>(cur, src, tgt, agg);
        sage_gemm_kernel<<<gemm_blocks, 256, 0, stream>>>(cur, agg, deg,
                                                          wh + (size_t)(2 * l) * WMAT_ELEMS, bl[l],
                                                          wh + (size_t)(2 * l + 1) * WMAT_ELEMS,
                                                          bufs[l & 1]);
        cur = bufs[l & 1];
    }

    // pooling + classifier
    zero_kernel<<<64, ZB, 0, stream>>>(g_sum, N_GRAPHS * C);
    zero_kernel<<<1, ZB, 0, stream>>>(g_cnt, N_GRAPHS);
    pool_kernel<<<(N_NODES * 32 + ZB - 1) / ZB, ZB, 0, stream>>>(cur, batch, g_sum, g_cnt);
    cls_kernel<<<N_GRAPHS, 64, 0, stream>>>(cur, root, g_sum, g_cnt,
                                            w_cls, b_cls, (float*)d_out);
}
